// AttentionLayer_72172630442215
// MI455X (gfx1250) — compile-verified, matrix operand order checked by reading, not run on hardware
//
#include <hip/hip_runtime.h>

// ---------------------------------------------------------------------------
// Causal attention layer for MI455X (gfx1250, wave32, WMMA + TDM).
// B=4, N=2048, DMODEL=DK=DV=1024.  f32 in/out, bf16 WMMA compute.
//
// Pipeline:
//   1) Q/K/V = x @ W^T         (NT GEMM, f32 -> bf16 convert in LDS staging)
//   2) E = (Q K^T) / sqrt(dk)  (NT GEMM on bf16; tiles staged by the Tensor
//                               Data Mover when available; causal tile skip)
//   3) P = causal softmax(E)   (rowwise, in-place f32, zeros masked region)
//   4) y = P @ V               (NN GEMM; V tile staged by TDM; causal k-stop)
// ---------------------------------------------------------------------------

typedef __attribute__((ext_vector_type(16))) __bf16 v16bf;
typedef __attribute__((ext_vector_type(8)))  float  v8f;
typedef __attribute__((ext_vector_type(4)))  __bf16 bf4;

typedef unsigned int u32x4 __attribute__((ext_vector_type(4)));
typedef int          i32x4 __attribute__((ext_vector_type(4)));
typedef int          i32x8 __attribute__((ext_vector_type(8)));

#define WMMA_BF16(a, b, c)                                                     \
  __builtin_amdgcn_wmma_f32_16x16x32_bf16(false, (a), false, (b), (short)0,    \
                                          (c), false, false)

#define LDT 40   // padded LDS row stride (elements) for 32-wide K tiles
#define LDV 136  // padded LDS row stride (elements) for 128-wide V tiles

#if defined(__has_builtin)
#if __has_builtin(__builtin_amdgcn_tensor_load_to_lds) &&                      \
    __has_builtin(__builtin_amdgcn_s_wait_tensorcnt)
#define USE_TDM 1
#endif
#endif
#ifndef USE_TDM
#define USE_TDM 0
#endif

#if USE_TDM
// Issue one TDM descriptor: load a [tile_h x tile_w] bf16 tile (row-major,
// row stride = stride_elems) from global memory into LDS, with hardware row
// padding so the LDS row stride becomes tile_w + 8 elements (16B pad).
// Encoding per CDNA5 ISA (D# groups 0/1).
static __device__ __forceinline__ void tdm_load_tile_bf16(
    const void* lds_dst, const void* gsrc, unsigned tile_w, unsigned tile_h,
    unsigned long long stride_elems, unsigned pad_interval_code,
    unsigned pad_amount_code) {
  const unsigned lds_addr = (unsigned)(size_t)lds_dst;  // low 32b = LDS offset
  const unsigned long long ga = (unsigned long long)(size_t)gsrc;
  const unsigned td0 = (unsigned)stride_elems;  // tensor_dim0 >= tile_w
  const unsigned td1 = 0x00100000u;             // tensor_dim1: ample

  u32x4 g0;
  g0[0] = 1u;                                            // count=1, user D#
  g0[1] = lds_addr;                                      // lds byte address
  g0[2] = (unsigned)ga;                                  // global addr [31:0]
  g0[3] = (unsigned)((ga >> 32) & 0x01FFFFFFu) | (2u << 30);  // [56:32]|type=2

  i32x8 g1;
  g1[0] = (int)((1u << 16)                   // data_size = 2 bytes
                | (1u << 20)                 // pad_enable
                | (pad_interval_code << 22)  // DWORDs between pads
                | (pad_amount_code << 25));  // pad amount (DWORDs)
  g1[1] = (int)((td0 & 0xFFFFu) << 16);                         // dim0 lo16
  g1[2] = (int)(((td0 >> 16) & 0xFFFFu) | ((td1 & 0xFFFFu) << 16));
  g1[3] = (int)(((td1 >> 16) & 0xFFFFu) | ((tile_w & 0xFFFFu) << 16));
  g1[4] = (int)(tile_h & 0xFFFFu);           // tile_dim1 (tile_dim2 = 0 -> 2D)
  g1[5] = (int)(unsigned)stride_elems;       // tensor_dim0_stride [31:0]
  g1[6] = (int)((unsigned)(stride_elems >> 32) & 0xFFFFu);
  g1[7] = 0;

  const i32x4 z4 = {0, 0, 0, 0};
#if __clang_major__ >= 23
  const i32x8 z8 = {0, 0, 0, 0, 0, 0, 0, 0};
  __builtin_amdgcn_tensor_load_to_lds(g0, g1, z4, z4, z8, 0);
#else
  __builtin_amdgcn_tensor_load_to_lds(g0, g1, z4, z4, 0);
#endif
}
#endif  // USE_TDM

// A-style fragment: source is row-major with the contraction dim contiguous.
// 16-bit A 16x32 layout (ISA 7.12.2): lanes 0-15 hold K 0-7 / 16-23,
// lanes 16-31 hold K 8-15 / 24-31, row M = lane % 16.
static __device__ __forceinline__ v16bf frag_k(const __bf16* p, int ld) {
  const int lane = threadIdx.x & 31;
  const int r    = lane & 15;
  const int kb   = (lane >> 4) << 3;
  const __bf16* q = p + r * ld + kb;
  v16bf f;
#pragma unroll
  for (int i = 0; i < 8; ++i) f[i] = q[i];
#pragma unroll
  for (int i = 0; i < 8; ++i) f[i + 8] = q[16 + i];
  return f;
}

// B-style fragment for NN GEMM: source tile is [K][N] row-major (N contiguous),
// lane holds column n = lane % 16, same per-lane K mapping as frag_k.
static __device__ __forceinline__ v16bf frag_n(const __bf16* p, int ld) {
  const int lane = threadIdx.x & 31;
  const int n    = lane & 15;
  const int kb   = (lane >> 4) << 3;
  v16bf f;
#pragma unroll
  for (int i = 0; i < 8; ++i) f[i] = p[(kb + i) * ld + n];
#pragma unroll
  for (int i = 0; i < 8; ++i) f[i + 8] = p[(16 + kb + i) * ld + n];
  return f;
}

static __device__ __forceinline__ v8f v8f_zero() {
  v8f z;
#pragma unroll
  for (int e = 0; e < 8; ++e) z[e] = 0.0f;
  return z;
}

// ---------------------------------------------------------------------------
// 1) Y[m,n] (bf16) = sum_d X[m,d] * W[n,d] ;  X:[M,K] f32, W:[N,K] f32
//    128x128 tile per block, 8 waves, each wave = 64x32 region (4x2 subtiles).
//    f32 -> bf16 conversion happens in the staging path (TDM cannot convert).
// ---------------------------------------------------------------------------
__global__ __launch_bounds__(256) void qkv_gemm_nt(
    const float* __restrict__ X, const float* __restrict__ W,
    __bf16* __restrict__ Y, int M, int N, int K) {
  __shared__ __align__(16) __bf16 lA[128][LDT];
  __shared__ __align__(16) __bf16 lB[128][LDT];

  const int tid = threadIdx.x;
  const int m0 = blockIdx.y * 128;
  const int n0 = blockIdx.x * 128;
  const int w  = tid >> 5;
  const int wr = w >> 2;   // 0..1 : 64 rows each
  const int wc = w & 3;    // 0..3 : 32 cols each

  v8f acc[4][2];
#pragma unroll
  for (int i = 0; i < 4; ++i)
#pragma unroll
    for (int j = 0; j < 2; ++j) acc[i][j] = v8f_zero();

  for (int k0 = 0; k0 < K; k0 += 32) {
#pragma unroll
    for (int p = 0; p < 4; ++p) {
      const int row = (tid >> 3) + p * 32;
      const int c   = (tid & 7) * 4;
      const float4 xa = *(const float4*)(X + (size_t)(m0 + row) * K + k0 + c);
      const float4 wb = *(const float4*)(W + (size_t)(n0 + row) * K + k0 + c);
      lA[row][c + 0] = (__bf16)xa.x; lA[row][c + 1] = (__bf16)xa.y;
      lA[row][c + 2] = (__bf16)xa.z; lA[row][c + 3] = (__bf16)xa.w;
      lB[row][c + 0] = (__bf16)wb.x; lB[row][c + 1] = (__bf16)wb.y;
      lB[row][c + 2] = (__bf16)wb.z; lB[row][c + 3] = (__bf16)wb.w;
    }
    __syncthreads();

    v16bf fa[4], fb[2];
#pragma unroll
    for (int i = 0; i < 4; ++i) fa[i] = frag_k(&lA[wr * 64 + i * 16][0], LDT);
#pragma unroll
    for (int j = 0; j < 2; ++j) fb[j] = frag_k(&lB[wc * 32 + j * 16][0], LDT);
#pragma unroll
    for (int i = 0; i < 4; ++i)
#pragma unroll
      for (int j = 0; j < 2; ++j) acc[i][j] = WMMA_BF16(fa[i], fb[j], acc[i][j]);
    __syncthreads();
  }

  const int lane = tid & 31;
  const int ln = lane & 15, lh = lane >> 4;
#pragma unroll
  for (int i = 0; i < 4; ++i)
#pragma unroll
    for (int j = 0; j < 2; ++j) {
      const int n = n0 + wc * 32 + j * 16 + ln;
#pragma unroll
      for (int r = 0; r < 8; ++r) {
        const int m = m0 + wr * 64 + i * 16 + r + 8 * lh;
        Y[(size_t)m * N + n] = (__bf16)acc[i][j][r];
      }
    }
}

// ---------------------------------------------------------------------------
// 2) E[q,s] = scale * sum_k Q[q,k] * K[s,k]   (per batch; bf16 in, f32 out)
//    Tiles staged by the Tensor Data Mover (wave 0 issues, all waves consume).
//    Blocks strictly above the causal diagonal exit immediately.
// ---------------------------------------------------------------------------
__global__ __launch_bounds__(256) void qk_gemm(
    const __bf16* __restrict__ Qb, const __bf16* __restrict__ Kb,
    float* __restrict__ E, int Nseq, int Dk, float scale) {
  const int qt = blockIdx.y, st = blockIdx.x;
  if (st > qt) return;  // fully masked tile

  __shared__ __align__(16) __bf16 lA[128][LDT];
  __shared__ __align__(16) __bf16 lB[128][LDT];

  const int b   = blockIdx.z;
  const int tid = threadIdx.x;
  const int q0 = qt * 128, s0 = st * 128;
  const __bf16* Qp = Qb + (size_t)b * Nseq * Dk;
  const __bf16* Kp = Kb + (size_t)b * Nseq * Dk;
  float* Ep = E + (size_t)b * Nseq * Nseq;
  const int w = tid >> 5, wr = w >> 2, wc = w & 3;

  v8f acc[4][2];
#pragma unroll
  for (int i = 0; i < 4; ++i)
#pragma unroll
    for (int j = 0; j < 2; ++j) acc[i][j] = v8f_zero();

  for (int k0 = 0; k0 < Dk; k0 += 32) {
#if USE_TDM
    if (tid < 32) {
      // 128 rows x 32 cols bf16, global row stride Dk, LDS rows padded to 40.
      // pad_interval code 3 -> every 16 DWORDs (one 64B row),
      // pad_amount  code 3 -> 4 DWORDs (16B) => LDS stride 80B = 40 elems.
      tdm_load_tile_bf16(&lA[0][0], Qp + (size_t)q0 * Dk + k0, 32, 128,
                         (unsigned long long)Dk, 3, 3);
      tdm_load_tile_bf16(&lB[0][0], Kp + (size_t)s0 * Dk + k0, 32, 128,
                         (unsigned long long)Dk, 3, 3);
      __builtin_amdgcn_s_wait_tensorcnt(0);
    }
#else
#pragma unroll
    for (int p = 0; p < 4; ++p) {
      const int row = (tid >> 3) + p * 32;
      const int c   = (tid & 7) * 4;
      *(bf4*)&lA[row][c] = *(const bf4*)(Qp + (size_t)(q0 + row) * Dk + k0 + c);
      *(bf4*)&lB[row][c] = *(const bf4*)(Kp + (size_t)(s0 + row) * Dk + k0 + c);
    }
#endif
    __syncthreads();

    v16bf fa[4], fb[2];
#pragma unroll
    for (int i = 0; i < 4; ++i) fa[i] = frag_k(&lA[wr * 64 + i * 16][0], LDT);
#pragma unroll
    for (int j = 0; j < 2; ++j) fb[j] = frag_k(&lB[wc * 32 + j * 16][0], LDT);
#pragma unroll
    for (int i = 0; i < 4; ++i)
#pragma unroll
      for (int j = 0; j < 2; ++j) acc[i][j] = WMMA_BF16(fa[i], fb[j], acc[i][j]);
    __syncthreads();
  }

  const int lane = tid & 31;
  const int ln = lane & 15, lh = lane >> 4;
#pragma unroll
  for (int i = 0; i < 4; ++i)
#pragma unroll
    for (int j = 0; j < 2; ++j) {
      const int s = s0 + wc * 32 + j * 16 + ln;
#pragma unroll
      for (int r = 0; r < 8; ++r) {
        const int q = q0 + wr * 64 + i * 16 + r + 8 * lh;
        Ep[(size_t)q * Nseq + s] = scale * acc[i][j][r];
      }
    }
}

// ---------------------------------------------------------------------------
// 3) Causal softmax over keys, one block per (b, q) row, in place (f32).
//    Writes zeros for s > q so the A*V GEMM can read dense rows.
// ---------------------------------------------------------------------------
__global__ __launch_bounds__(256) void softmax_causal(float* __restrict__ E,
                                                      int Nseq) {
  const int q = blockIdx.x, b = blockIdx.y;
  float* row = E + ((size_t)b * Nseq + q) * Nseq;
  const int len  = q + 1;
  const int tid  = threadIdx.x;
  const int lane = tid & 31, wid = tid >> 5;
  __shared__ float sred[8];

  float m = -3.402823466e38f;
  for (int i = tid; i < len; i += 256) m = fmaxf(m, row[i]);
#pragma unroll
  for (int o = 16; o > 0; o >>= 1) m = fmaxf(m, __shfl_xor(m, o, 32));
  if (lane == 0) sred[wid] = m;
  __syncthreads();
  if (tid == 0) {
    float v = sred[0];
#pragma unroll
    for (int i = 1; i < 8; ++i) v = fmaxf(v, sred[i]);
    sred[0] = v;
  }
  __syncthreads();
  m = sred[0];
  __syncthreads();

  float s = 0.0f;
  for (int i = tid; i < len; i += 256) s += __expf(row[i] - m);
#pragma unroll
  for (int o = 16; o > 0; o >>= 1) s += __shfl_xor(s, o, 32);
  if (lane == 0) sred[wid] = s;
  __syncthreads();
  if (tid == 0) {
    float v = 0.0f;
#pragma unroll
    for (int i = 0; i < 8; ++i) v += sred[i];
    sred[0] = v;
  }
  __syncthreads();
  const float inv = 1.0f / sred[0];

  for (int i = tid; i < len; i += 256) row[i] = __expf(row[i] - m) * inv;
  for (int i = len + tid; i < Nseq; i += 256) row[i] = 0.0f;
}

// ---------------------------------------------------------------------------
// 4) y[q,v] = sum_s P[q,s] * V[s,v]   (NN GEMM; P f32 staged->bf16, V bf16
//    staged by TDM).  P[q, s > q] == 0, so the k loop stops at the diagonal.
// ---------------------------------------------------------------------------
__global__ __launch_bounds__(256) void av_gemm(
    const float* __restrict__ P, const __bf16* __restrict__ Vb,
    float* __restrict__ Y, int Nseq, int Dv) {
  __shared__ __align__(16) __bf16 lA[128][LDT];
  __shared__ __align__(16) __bf16 lV[32][LDV];

  const int b   = blockIdx.z;
  const int qt  = blockIdx.y, vt = blockIdx.x;
  const int tid = threadIdx.x;
  const int q0 = qt * 128, v0 = vt * 128;
  const float*  Pp = P  + (size_t)b * Nseq * Nseq;
  const __bf16* Vp = Vb + (size_t)b * Nseq * Dv;
  float* Yp = Y + (size_t)b * Nseq * Dv;
  const int w = tid >> 5, wr = w >> 2, wc = w & 3;
  const int kmax = ((qt + 1) * 128 < Nseq) ? (qt + 1) * 128 : Nseq;

  v8f acc[4][2];
#pragma unroll
  for (int i = 0; i < 4; ++i)
#pragma unroll
    for (int j = 0; j < 2; ++j) acc[i][j] = v8f_zero();

  for (int k0 = 0; k0 < kmax; k0 += 32) {
#if USE_TDM
    if (tid < 32) {
      // V tile: 32 rows x 128 cols bf16, global row stride Dv.
      // pad_interval code 5 -> every 64 DWORDs (one 256B row),
      // pad_amount  code 3 -> 4 DWORDs (16B) => LDS stride 272B = 136 elems.
      tdm_load_tile_bf16(&lV[0][0], Vp + (size_t)k0 * Dv + v0, 128, 32,
                         (unsigned long long)Dv, 5, 3);
      __builtin_amdgcn_s_wait_tensorcnt(0);
    }
#else
#pragma unroll
    for (int p = 0; p < 4; ++p) {
      const int vr = (tid >> 5) + p * 8;
      const int vc = (tid & 31) * 4;
      *(bf4*)&lV[vr][vc] = *(const bf4*)(Vp + (size_t)(k0 + vr) * Dv + v0 + vc);
    }
#endif
    // A: P rows q0..q0+127, cols k0..k0+31 (f32 -> bf16, manual staging)
#pragma unroll
    for (int p = 0; p < 4; ++p) {
      const int row = (tid >> 3) + p * 32;
      const int c   = (tid & 7) * 4;
      const float4 pa =
          *(const float4*)(Pp + (size_t)(q0 + row) * Nseq + k0 + c);
      lA[row][c + 0] = (__bf16)pa.x; lA[row][c + 1] = (__bf16)pa.y;
      lA[row][c + 2] = (__bf16)pa.z; lA[row][c + 3] = (__bf16)pa.w;
    }
    __syncthreads();

    v16bf fa[4], fb[2];
#pragma unroll
    for (int i = 0; i < 4; ++i) fa[i] = frag_k(&lA[wr * 64 + i * 16][0], LDT);
#pragma unroll
    for (int j = 0; j < 2; ++j) fb[j] = frag_n(&lV[0][wc * 32 + j * 16], LDV);
#pragma unroll
    for (int i = 0; i < 4; ++i)
#pragma unroll
      for (int j = 0; j < 2; ++j) acc[i][j] = WMMA_BF16(fa[i], fb[j], acc[i][j]);
    __syncthreads();
  }

  const int lane = tid & 31;
  const int ln = lane & 15, lh = lane >> 4;
#pragma unroll
  for (int i = 0; i < 4; ++i)
#pragma unroll
    for (int j = 0; j < 2; ++j) {
      const int v = v0 + wc * 32 + j * 16 + ln;
#pragma unroll
      for (int r = 0; r < 8; ++r) {
        const int q = q0 + wr * 64 + i * 16 + r + 8 * lh;
        Yp[(size_t)q * Dv + v] = acc[i][j][r];
      }
    }
}

// ---------------------------------------------------------------------------
// Host-side launcher
// ---------------------------------------------------------------------------
extern "C" void kernel_launch(void* const* d_in, const int* in_sizes, int n_in,
                              void* d_out, int out_size, void* d_ws,
                              size_t ws_size, hipStream_t stream) {
  (void)in_sizes; (void)n_in; (void)out_size; (void)ws_size;
  const float* x  = (const float*)d_in[0];
  const float* Wq = (const float*)d_in[1];
  const float* Wk = (const float*)d_in[2];
  const float* Wv = (const float*)d_in[3];
  float* out = (float*)d_out;

  const int B = 4, N = 2048, D = 1024;
  const size_t rows = (size_t)B * N;  // 8192

  // Workspace layout: Qb/Kb/Vb bf16 (3 x 16 MB), E f32 (64 MB) -> 112 MB.
  __bf16* Qb = (__bf16*)d_ws;
  __bf16* Kb = Qb + rows * D;
  __bf16* Vb = Kb + rows * D;
  float*  E  = (float*)(Vb + rows * D);

  dim3 blk(256);

  dim3 gProj(D / 128, (unsigned)(rows / 128));  // (8, 64)
  qkv_gemm_nt<<<gProj, blk, 0, stream>>>(x, Wq, Qb, (int)rows, D, D);
  qkv_gemm_nt<<<gProj, blk, 0, stream>>>(x, Wk, Kb, (int)rows, D, D);
  qkv_gemm_nt<<<gProj, blk, 0, stream>>>(x, Wv, Vb, (int)rows, D, D);

  dim3 gQK(N / 128, N / 128, B);  // (s tiles, q tiles, batch)
  qk_gemm<<<gQK, blk, 0, stream>>>(Qb, Kb, E, N, D, 0.03125f /* 1/sqrt(1024) */);

  softmax_causal<<<dim3(N, B), blk, 0, stream>>>(E, N);

  dim3 gAV(D / 128, N / 128, B);  // (v tiles, q tiles, batch)
  av_gemm<<<gAV, blk, 0, stream>>>(E, Vb, out, N, D);
}
